// HierarchicalGTTFN_64888365907995
// MI455X (gfx1250) — matrix operand order repaired and checked by reading, not compiled, not run on hardware
//
#include <hip/hip_runtime.h>
#include <hip/hip_bf16.h>

typedef __attribute__((ext_vector_type(16))) _Float16 v16h;
typedef __attribute__((ext_vector_type(8)))  _Float16 v8h;
typedef __attribute__((ext_vector_type(8)))  float    v8f;

#define NB    8
#define N0    2048
#define M1    512
#define M2    128
#define KNBR  16
#define CCH   32
#define NRBF  32
#define RH    64
#define NCLS  40

#define FRAG_H 512           // halfs per packed B fragment (32 lanes x 16)
#define NWFRAG 28            // fragments per GTTFN layer weight blob
#define WBLOB_H (NWFRAG * FRAG_H)
#define WBLOB_B (WBLOB_H * 2)

// ---------------------------------------------------------------- WMMA helpers
__device__ __forceinline__ v8f wmma_f16(v16h a, v16h b, v8f c) {
  // D = A(16x32 f16) x B(32x16 f16) + C(16x16 f32)
  return __builtin_amdgcn_wmma_f32_16x16x32_f16(false, a, false, b, (short)0, c,
                                                false, false);
}

// A fragment: 16x32 f16, row-major matrix with leading dim ldm (in halfs).
// Lane L holds row m = L%16; halfs {c0..c0+7} and {c0+16..c0+23}, c0 = 8*(L>=16).
__device__ __forceinline__ v16h frag_a(const _Float16* mat, int ldm) {
  int lane = threadIdx.x & 31;
  const _Float16* p = mat + (size_t)(lane & 15) * ldm + ((lane >> 4) << 3);
  v8h lo = *(const v8h*)(p);
  v8h hi = *(const v8h*)(p + 16);
  v16h a;
#pragma unroll
  for (int e = 0; e < 8; ++e) { a[e] = lo[e]; a[8 + e] = hi[e]; }
  return a;
}

// A fragment with per-row gather (row index per m from nbr16[])
__device__ __forceinline__ v16h frag_a_gather(const _Float16* base,
                                              const int* nbr16, int ldm) {
  int lane = threadIdx.x & 31;
  int row = nbr16[lane & 15];
  const _Float16* p = base + (size_t)row * ldm + ((lane >> 4) << 3);
  v8h lo = *(const v8h*)(p);
  v8h hi = *(const v8h*)(p + 16);
  v16h a;
#pragma unroll
  for (int e = 0; e < 8; ++e) { a[e] = lo[e]; a[8 + e] = hi[e]; }
  return a;
}

// B fragment from a packed fragment-major blob: blob[frag][lane][16 halfs].
// One contiguous 32B read per lane (2 x b128).
__device__ __forceinline__ v16h frag_b_packed(const _Float16* blob, int frag) {
  int lane = threadIdx.x & 31;
  return *(const v16h*)(blob + (size_t)frag * FRAG_H + lane * 16);
}

// Sum over the 16 m-rows of a D tile; result valid in all lanes (col n = lane&15).
__device__ __forceinline__ float dtile_colsum(v8f d) {
  float s = 0.f;
#pragma unroll
  for (int r = 0; r < 8; ++r) s += d[r];
  s += __shfl_xor(s, 16, 32);
  return s;
}

__device__ __forceinline__ float silu_f(float x) { return x / (1.f + __expf(-x)); }
__device__ __forceinline__ float sigm_f(float x) { return 1.f / (1.f + __expf(-x)); }

// ---------------------------------------------------------------- TDM staging
#if __has_builtin(__builtin_amdgcn_tensor_load_to_lds) && \
    __has_builtin(__builtin_amdgcn_s_wait_tensorcnt)
#define HG_HAS_TDM 1
typedef __attribute__((ext_vector_type(4))) unsigned hg_u32x4;
typedef __attribute__((ext_vector_type(8))) int      hg_i32x8;
typedef __attribute__((ext_vector_type(4))) int      hg_i32x4;

// DMA a contiguous global blob (n8 x 8-byte elements) into LDS via the Tensor
// Data Mover (1-D tile descriptor), then wait for TENSORcnt.
// This toolchain's builtin is the 6-arg form:
//   (u32x4 g0, i32x8 g1, i32x4 g2, i32x4 g3, i32x8, i32 cpol)
__device__ __forceinline__ void hg_tdm_load(const void* gsrc, void* lds_dst,
                                            unsigned n8) {
  unsigned long long ga = (unsigned long long)(uintptr_t)gsrc;
  unsigned la = (unsigned)(uintptr_t)lds_dst;   // low 32 bits = LDS byte offset
  hg_u32x4 g0;
  g0[0] = 1u;                                   // count=1 valid, user mode
  g0[1] = la;                                   // lds_addr
  g0[2] = (unsigned)(ga & 0xFFFFFFFFu);         // global_addr[31:0]
  g0[3] = ((unsigned)(ga >> 32) & 0x01FFFFFFu)  // global_addr[56:32]
          | (2u << 30);                         // type = 2 ("image")
  hg_i32x8 g1;
  g1[0] = (int)(3u << 16);                      // data_size=8B, no multicast
  g1[1] = (int)((n8 & 0xFFFFu) << 16);          // tensor_dim0[15:0]
  g1[2] = (int)((n8 >> 16) | (1u << 16));       // tensor_dim0[31:16], dim1=1
  g1[3] = (int)((n8 & 0xFFFFu) << 16);          // tile_dim0 = n8
  g1[4] = 0;                                    // tile_dim1/2 unused
  g1[5] = (int)n8;                              // tensor_dim0_stride
  g1[6] = 0;
  g1[7] = 0;
  hg_i32x4 gz4 = {0, 0, 0, 0};
  hg_i32x8 gz8 = {0, 0, 0, 0, 0, 0, 0, 0};
  __builtin_amdgcn_tensor_load_to_lds(g0, g1, gz4, gz4, gz8, 0);
  __builtin_amdgcn_s_wait_tensorcnt(0);
}
#else
#define HG_HAS_TDM 0
#endif

// ---------------------------------------------------------------- param pack
// Pack a (K x ldw) row-major f32 weight matrix into f16 B fragments.
// Fragment f covers cols n = f*16 + lane%16, rows k = k0 + 8*(lane/16) + e (e<8), k0 + 16 + 8*(lane/16) + e-8 (e>=8).
// Rows >= valid_rows are zero-padded (handles the cin=1 init layer).
__global__ void hg_pack_b(const float* __restrict__ W, int valid_rows, int ldw,
                          int k0, int nblocks, _Float16* __restrict__ dst) {
  int gid = blockIdx.x * blockDim.x + threadIdx.x;
  if (gid >= nblocks * FRAG_H) return;
  int f = gid >> 9;
  int lane = (gid >> 4) & 31;
  int e = gid & 15;
  int n = f * 16 + (lane & 15);
  int k = k0 + ((e & 8) << 1) + ((lane >> 4) << 3) + (e & 7);
  float val = (k < valid_rows) ? W[(size_t)k * ldw + n] : 0.f;
  dst[gid] = (_Float16)val;
}

// ---------------------------------------------------------------- embed
__global__ void hg_init_embed(const float* __restrict__ pos, int NT,
                              float* __restrict__ sf, _Float16* __restrict__ sh,
                              float* __restrict__ vf, _Float16* __restrict__ vh) {
  int i = blockIdx.x * blockDim.x + threadIdx.x;
  if (i >= NT) return;
  size_t vs = (size_t)NT * CCH;
  float x = pos[i * 3], y = pos[i * 3 + 1], z = pos[i * 3 + 2];
  float nrm = sqrtf(x * x + y * y + z * z);
  float inv = 1.f / fmaxf(nrm, 1e-8f);
  float u[3] = {x * inv, y * inv, z * inv};
#pragma unroll 4
  for (int c = 0; c < CCH; ++c) {
    float sv = (c == 0) ? nrm : 0.f;
    sf[(size_t)i * CCH + c] = sv;
    sh[(size_t)i * CCH + c] = (_Float16)sv;
    for (int d = 0; d < 3; ++d) {
      float vv = (c == 0) ? u[d] : 0.f;
      vf[d * vs + (size_t)i * CCH + c] = vv;
      vh[d * vs + (size_t)i * CCH + c] = (_Float16)vv;
    }
  }
}

// ---------------------------------------------------------------- kNN + RBF
__global__ void hg_knn(const float* __restrict__ pts, int Nn, int NT,
                       _Float16* __restrict__ rbf, float* __restrict__ rhat,
                       int* __restrict__ nbr) {
  int gid = blockIdx.x * blockDim.x + threadIdx.x;
  if (gid >= NT) return;
  int b = gid / Nn, i = gid % Nn;
  const float* pb = pts + (size_t)b * Nn * 3;
  float xi = pb[i * 3], yi = pb[i * 3 + 1], zi = pb[i * 3 + 2];
  float bd[KNBR];
  int bi[KNBR];
#pragma unroll
  for (int t = 0; t < KNBR; ++t) { bd[t] = 3e30f; bi[t] = 0; }
  for (int j = 0; j < Nn; ++j) {
    if (j == i) continue;
    float dx = xi - pb[j * 3], dy = yi - pb[j * 3 + 1], dz = zi - pb[j * 3 + 2];
    float d2 = dx * dx + dy * dy + dz * dz;
    if (d2 < bd[KNBR - 1]) {
      float dn = d2; int jn = j;
#pragma unroll
      for (int t = 0; t < KNBR; ++t) {
        if (dn < bd[t]) {
          float td = bd[t]; int tj = bi[t];
          bd[t] = dn; bi[t] = jn; dn = td; jn = tj;
        }
      }
    }
  }
  for (int t = 0; t < KNBR; ++t) {
    int j = bi[t];
    nbr[(size_t)gid * KNBR + t] = b * Nn + j;
    float dx = xi - pb[j * 3], dy = yi - pb[j * 3 + 1], dz = zi - pb[j * 3 + 2];
    float dist = fmaxf(sqrtf(dx * dx + dy * dy + dz * dz), 1e-8f);
    float inv = 1.f / dist;
    size_t ro = ((size_t)gid * KNBR + t) * 3;
    rhat[ro + 0] = dx * inv; rhat[ro + 1] = dy * inv; rhat[ro + 2] = dz * inv;
    float dc = fminf(fmaxf(dist, 0.f), 1.f);
    float env = 0.5f * (__cosf(3.14159265358979f * dc) + 1.f);
    size_t bo = (size_t)gid * KNBR * NRBF + (size_t)t * NRBF;
#pragma unroll 8
    for (int r = 0; r < NRBF; ++r) {
      float mu = (float)r * (1.0f / (NRBF - 1));
      float u = (dist - mu) * (float)NRBF;  // /width, width = 1/NRBF
      rbf[bo + r] = (_Float16)(__expf(-u * u) * env);
    }
  }
}

// ---------------------------------------------------------------- FPS
__global__ __launch_bounds__(256) void hg_fps(const float* __restrict__ pts,
                                              int Nn, int M,
                                              float* __restrict__ cents) {
  __shared__ float dist[2048];
  __shared__ float rmax[256];
  __shared__ int rarg[256];
  __shared__ int s_cur;
  int b = blockIdx.x, tid = threadIdx.x;
  const float* pb = pts + (size_t)b * Nn * 3;
  for (int i = tid; i < Nn; i += 256) dist[i] = 3e30f;
  if (tid == 0) s_cur = 0;
  __syncthreads();
  for (int m = 0; m < M; ++m) {
    int cur = s_cur;
    float cx = pb[cur * 3], cy = pb[cur * 3 + 1], cz = pb[cur * 3 + 2];
    if (tid == 0) {
      size_t o = ((size_t)b * M + m) * 3;
      cents[o] = cx; cents[o + 1] = cy; cents[o + 2] = cz;
    }
    float best = -3e30f; int barg = 0;
    for (int i = tid; i < Nn; i += 256) {
      float dx = pb[i * 3] - cx, dy = pb[i * 3 + 1] - cy, dz = pb[i * 3 + 2] - cz;
      float nd = fminf(dist[i], dx * dx + dy * dy + dz * dz);
      dist[i] = nd;
      if (nd > best) { best = nd; barg = i; }
    }
    rmax[tid] = best; rarg[tid] = barg;
    __syncthreads();
    for (int s = 128; s > 0; s >>= 1) {
      if (tid < s && rmax[tid + s] > rmax[tid]) {
        rmax[tid] = rmax[tid + s]; rarg[tid] = rarg[tid + s];
      }
      __syncthreads();
    }
    if (tid == 0) s_cur = rarg[0];
    __syncthreads();
  }
}

// ---------------------------------------------------------------- ball query
__global__ void hg_ballq(const float* __restrict__ src, int Nsrc,
                         const float* __restrict__ cents, int M, int total,
                         float r2, int* __restrict__ bidx) {
  int gid = blockIdx.x * blockDim.x + threadIdx.x;
  if (gid >= total) return;
  int b = gid / M;
  const float* pb = src + (size_t)b * Nsrc * 3;
  float cx = cents[gid * 3], cy = cents[gid * 3 + 1], cz = cents[gid * 3 + 2];
  float bd[KNBR];
  int bi[KNBR];
#pragma unroll
  for (int t = 0; t < KNBR; ++t) { bd[t] = 3e30f; bi[t] = 0; }
  for (int j = 0; j < Nsrc; ++j) {
    float dx = cx - pb[j * 3], dy = cy - pb[j * 3 + 1], dz = cz - pb[j * 3 + 2];
    float d2 = dx * dx + dy * dy + dz * dz;
    if (d2 <= r2 && d2 < bd[KNBR - 1]) {
      float dn = d2; int jn = j;
#pragma unroll
      for (int t = 0; t < KNBR; ++t) {
        if (dn < bd[t]) {
          float td = bd[t]; int tj = bi[t];
          bd[t] = dn; bi[t] = jn; dn = td; jn = tj;
        }
      }
    }
  }
  int j0 = bi[0];  // center itself (d2 = 0) is always present
  for (int t = 0; t < KNBR; ++t) {
    int j = (bd[t] < 1e29f) ? bi[t] : j0;
    bidx[(size_t)gid * KNBR + t] = b * Nsrc + j;
  }
}

// ---------------------------------------------------------------- max pool
__global__ void hg_pool(const int* __restrict__ bidx, int NTdst,
                        const float* __restrict__ sfi, const float* __restrict__ vfi,
                        size_t vsi, float* __restrict__ sfo,
                        _Float16* __restrict__ sho, float* __restrict__ vfo,
                        _Float16* __restrict__ vho, size_t vso) {
  int gid = blockIdx.x * blockDim.x + threadIdx.x;
  if (gid >= NTdst * CCH) return;
  int node = gid >> 5, c = gid & 31;
  const int* nb = bidx + (size_t)node * KNBR;
  float sm = -3e30f, v0 = -3e30f, v1 = -3e30f, v2 = -3e30f;
#pragma unroll 4
  for (int t = 0; t < KNBR; ++t) {
    size_t row = (size_t)nb[t] * CCH + c;
    sm = fmaxf(sm, sfi[row]);
    v0 = fmaxf(v0, vfi[row]);
    v1 = fmaxf(v1, vfi[vsi + row]);
    v2 = fmaxf(v2, vfi[2 * vsi + row]);
  }
  size_t o = (size_t)node * CCH + c;
  sfo[o] = sm; sho[o] = (_Float16)sm;
  vfo[o] = v0; vho[o] = (_Float16)v0;
  vfo[vso + o] = v1; vho[vso + o] = (_Float16)v1;
  vfo[2 * vso + o] = v2; vho[2 * vso + o] = (_Float16)v2;
}

// ---------------------------------------------------------------- channel mixer
// Weights come pre-packed in fragment-major layout: [Ms f0, Ms f1, Mv f0, Mv f1]
__global__ __launch_bounds__(256) void hg_mixer(
    int NT, const _Float16* __restrict__ xs, const _Float16* __restrict__ xv,
    const _Float16* __restrict__ mblob,
    float* __restrict__ ysf, _Float16* __restrict__ ysh,
    float* __restrict__ yvf, _Float16* __restrict__ yvh) {
  int wave = threadIdx.x >> 5, lane = threadIdx.x & 31;
  size_t tile = (size_t)blockIdx.x * 8 + wave;
  size_t rbase = tile * 16;
  size_t vs = (size_t)NT * CCH;
  int mb = (lane >> 4) << 3, nn = lane & 15;
  v16h bs[2] = {frag_b_packed(mblob, 0), frag_b_packed(mblob, 1)};
  v16h bv[2] = {frag_b_packed(mblob, 2), frag_b_packed(mblob, 3)};
  {
    v16h a = frag_a(xs + rbase * CCH, CCH);
#pragma unroll
    for (int t = 0; t < 2; ++t) {
      v8f c = {};
      c = wmma_f16(a, bs[t], c);
#pragma unroll
      for (int r = 0; r < 8; ++r) {
        size_t o = (rbase + mb + r) * CCH + t * 16 + nn;
        ysf[o] = c[r]; ysh[o] = (_Float16)c[r];
      }
    }
  }
#pragma unroll
  for (int d = 0; d < 3; ++d) {
    v16h a = frag_a(xv + d * vs + rbase * CCH, CCH);
#pragma unroll
    for (int t = 0; t < 2; ++t) {
      v8f c = {};
      c = wmma_f16(a, bv[t], c);
#pragma unroll
      for (int r = 0; r < 8; ++r) {
        size_t o = d * vs + (rbase + mb + r) * CCH + t * 16 + nn;
        yvf[o] = c[r]; yvh[o] = (_Float16)c[r];
      }
    }
  }
}

// ---------------------------------------------------------------- GTTFN layer
// One wave per node; 8 waves per block; 32 WMMAs per node. The layer's 28
// packed B fragments (28KB) are DMA'd into LDS once per block via the TDM.
// Blob layout: W1 f0-3 | W2(k0-31) f4-11 | W2(k32-63) f12-19 | W00 f20-21 |
//              W11 f22-23 | W01 f24-25 | W10 f26-27
__global__ __launch_bounds__(256) void hg_gttfn(
    int NT, int residual,
    const _Float16* __restrict__ s_h, const float* __restrict__ s_f,
    const _Float16* __restrict__ v_h, const float* __restrict__ v_f,
    const _Float16* __restrict__ rbf_h, const float* __restrict__ rhat,
    const int* __restrict__ nbr, const _Float16* __restrict__ wblob,
    const float* __restrict__ b1, const float* __restrict__ b2,
    const float* __restrict__ Wg, const float* __restrict__ bg,
    float* __restrict__ so_f, _Float16* __restrict__ so_h,
    float* __restrict__ vo_f, _Float16* __restrict__ vo_h) {
  __shared__ alignas(16) _Float16 sh_wb[WBLOB_H];
  __shared__ alignas(16) _Float16 sh_h[8][16 * RH];
  __shared__ alignas(16) _Float16 sh_dot[8][16 * CCH];
  __shared__ float sh_sout[8][CCH];
  __shared__ float sh_rhat[8][KNBR * 3];

  const int wave = threadIdx.x >> 5;
  const int lane = threadIdx.x & 31;
  const int n = blockIdx.x * 8 + wave;  // grid sized exactly
  const size_t vs = (size_t)NT * CCH;
  const int nn = lane & 15;
  const int mb = (lane >> 4) << 3;

  // stage packed weights into LDS (once per block)
#if HG_HAS_TDM
  if (threadIdx.x < 32) hg_tdm_load(wblob, sh_wb, WBLOB_B / 8);
#else
  for (int i = threadIdx.x; i < WBLOB_B / 16; i += 256)
    ((uint4*)sh_wb)[i] = ((const uint4*)wblob)[i];
#endif

  // stage r_hat for this node (48 floats)
  {
    int i = lane;
    if (i < KNBR * 3) sh_rhat[wave][i] = rhat[(size_t)n * KNBR * 3 + i];
    i = lane + 32;
    if (i < KNBR * 3) sh_rhat[wave][i] = rhat[(size_t)n * KNBR * 3 + i];
  }
  __syncthreads();  // sh_wb + sh_rhat visible

  // ---- radial MLP layer 1: h = silu(rbf @ W1 + b1)   (16x64)
  v16h a_rbf = frag_a(rbf_h + (size_t)n * (KNBR * NRBF), NRBF);
#pragma unroll
  for (int t = 0; t < RH / 16; ++t) {
    v8f c = {};
    c = wmma_f16(a_rbf, frag_b_packed(sh_wb, t), c);
    float bias = b1[t * 16 + nn];
#pragma unroll
    for (int r = 0; r < 8; ++r) {
      float x = silu_f(c[r] + bias);
      sh_h[wave][(mb + r) * RH + t * 16 + nn] = (_Float16)x;
    }
  }
  __syncthreads();  // sh_h visible

  // ---- radial MLP layer 2: w = h @ W2 + b2           (16x128)
  v16h a_hlo = frag_a(sh_h[wave], RH);
  v16h a_hhi = frag_a(sh_h[wave] + 32, RH);
  v8f wt[8];  // [0,1]=w00 [2,3]=w11 [4,5]=w01 [6,7]=w10
#pragma unroll
  for (int t = 0; t < 8; ++t) {
    v8f c = {};
    c = wmma_f16(a_hlo, frag_b_packed(sh_wb, 4 + t), c);
    c = wmma_f16(a_hhi, frag_b_packed(sh_wb, 12 + t), c);
    float bias = b2[t * 16 + nn];
#pragma unroll
    for (int r = 0; r < 8; ++r) c[r] += bias;
    wt[t] = c;
  }

  // gathered s_j fragment (rows = neighbor feature rows)
  const int* nbrn = nbr + (size_t)n * KNBR;
  v16h a_sj = frag_a_gather(s_h, nbrn, CCH);

  // dot[k,c] = sum_d v_j[k,c,d]*rhat[k,d] -> staged as A operand
  {
    int k = nn;
    int c0 = (lane >> 4) << 4;
    int j = nbrn[k];
    float r0 = sh_rhat[wave][k * 3 + 0];
    float r1 = sh_rhat[wave][k * 3 + 1];
    float r2 = sh_rhat[wave][k * 3 + 2];
    const float* vrow = v_f + (size_t)j * CCH;
#pragma unroll
    for (int cc = 0; cc < 16; ++cc) {
      int c = c0 + cc;
      float dv = vrow[c] * r0 + vrow[vs + c] * r1 + vrow[2 * vs + c] * r2;
      sh_dot[wave][k * CCH + c] = (_Float16)dv;
    }
  }
  __syncthreads();  // sh_dot visible
  v16h a_dot = frag_a(sh_dot[wave], CCH);

  // ---- scalar path: s_out = mean_k(m00 + m11)
  float s_col[2];
#pragma unroll
  for (int t = 0; t < 2; ++t) {
    v8f d00 = {};
    d00 = wmma_f16(a_sj, frag_b_packed(sh_wb, 20 + t), d00);
    v8f d11 = {};
    d11 = wmma_f16(a_dot, frag_b_packed(sh_wb, 22 + t), d11);
    v8f m = {};
#pragma unroll
    for (int r = 0; r < 8; ++r)
      m[r] = d00[r] * wt[t][r] + d11[r] * wt[2 + t][r];
    s_col[t] = dtile_colsum(m) * (1.f / KNBR);
  }
  if (lane < 16) {
    sh_sout[wave][lane] = s_col[0];
    sh_sout[wave][16 + lane] = s_col[1];
  }

  // m01 tiles (kept for the vector path)
  v8f d01[2];
#pragma unroll
  for (int t = 0; t < 2; ++t) {
    v8f c = {};
    c = wmma_f16(a_sj, frag_b_packed(sh_wb, 24 + t), c);
#pragma unroll
    for (int r = 0; r < 8; ++r) c[r] *= wt[4 + t][r];
    d01[t] = c;
  }
  __syncthreads();  // sh_sout visible

  // ---- vector path: v_out[c,d] = mean_k(m01[k,c]*rhat[k,d] + m10_d[k,c])
  v16h b10[2] = {frag_b_packed(sh_wb, 26), frag_b_packed(sh_wb, 27)};
  float v_col[2][3];
#pragma unroll
  for (int d = 0; d < 3; ++d) {
    v16h a_vd = frag_a_gather(v_h + (size_t)d * vs, nbrn, CCH);
#pragma unroll
    for (int t = 0; t < 2; ++t) {
      v8f m10 = {};
      m10 = wmma_f16(a_vd, b10[t], m10);
      float acc = 0.f;
#pragma unroll
      for (int r = 0; r < 8; ++r) {
        int k = mb + r;
        acc += m10[r] * wt[6 + t][r] + d01[t][r] * sh_rhat[wave][k * 3 + d];
      }
      acc += __shfl_xor(acc, 16, 32);
      v_col[t][d] = acc * (1.f / KNBR);
    }
  }

  // ---- gate, activation, residual, writeback (lanes 0..15 own 2 channels)
  if (lane < 16) {
#pragma unroll
    for (int t = 0; t < 2; ++t) {
      int c = t * 16 + lane;
      float g = bg[c];
#pragma unroll 8
      for (int cc = 0; cc < CCH; ++cc) g += sh_sout[wave][cc] * Wg[cc * CCH + c];
      g = sigm_f(g);
      float sn = silu_f(s_col[t]);
      size_t o = (size_t)n * CCH + c;
      if (residual) sn += s_f[o];
      so_f[o] = sn; so_h[o] = (_Float16)sn;
#pragma unroll
      for (int d = 0; d < 3; ++d) {
        float vn = v_col[t][d] * g;
        if (residual) vn += v_f[d * vs + o];
        vo_f[d * vs + o] = vn; vo_h[d * vs + o] = (_Float16)vn;
      }
    }
  }
}

// ---------------------------------------------------------------- descriptor
__global__ void hg_desc(const float* __restrict__ sf, const float* __restrict__ vf,
                        size_t vs, int Mn, float* __restrict__ desc) {
  int b = blockIdx.x, c = threadIdx.x;  // 64 threads
  float best = -3e30f;
  for (int m = 0; m < Mn; ++m) {
    size_t row = (size_t)(b * Mn + m) * CCH;
    float val;
    if (c < CCH) {
      val = sf[row + c];
    } else {
      int ch = c - CCH;
      float a0 = vf[row + ch], a1 = vf[vs + row + ch], a2 = vf[2 * vs + row + ch];
      val = sqrtf(a0 * a0 + a1 * a1 + a2 * a2);
    }
    best = fmaxf(best, val);
  }
  desc[b * 2 * CCH + c] = best;
}

// ---------------------------------------------------------------- MLP head
__global__ __launch_bounds__(128) void hg_head(
    const float* __restrict__ desc, const float* __restrict__ W0,
    const float* __restrict__ b0, const float* __restrict__ g0,
    const float* __restrict__ be0, const float* __restrict__ W1,
    const float* __restrict__ b1, const float* __restrict__ g1,
    const float* __restrict__ be1, const float* __restrict__ W2,
    const float* __restrict__ b2, float* __restrict__ out) {
  __shared__ float h0[128], h1[64], red[2];
  int tid = threadIdx.x;
  for (int b = 0; b < NB; ++b) {
    {
      float acc = b0[tid];
      for (int c = 0; c < 2 * CCH; ++c) acc += desc[b * 2 * CCH + c] * W0[c * 128 + tid];
      h0[tid] = silu_f(acc);
    }
    __syncthreads();
    if (tid == 0) {
      float mu = 0.f;
      for (int i = 0; i < 128; ++i) mu += h0[i];
      mu *= (1.f / 128.f);
      float var = 0.f;
      for (int i = 0; i < 128; ++i) { float d = h0[i] - mu; var += d * d; }
      red[0] = mu; red[1] = var * (1.f / 128.f);
    }
    __syncthreads();
    float n0v = (h0[tid] - red[0]) * rsqrtf(red[1] + 1e-5f) * g0[tid] + be0[tid];
    __syncthreads();
    h0[tid] = n0v;
    __syncthreads();
    if (tid < 64) {
      float acc = b1[tid];
      for (int c = 0; c < 128; ++c) acc += h0[c] * W1[c * 64 + tid];
      h1[tid] = silu_f(acc);
    }
    __syncthreads();
    if (tid == 0) {
      float mu = 0.f;
      for (int i = 0; i < 64; ++i) mu += h1[i];
      mu *= (1.f / 64.f);
      float var = 0.f;
      for (int i = 0; i < 64; ++i) { float d = h1[i] - mu; var += d * d; }
      red[0] = mu; red[1] = var * (1.f / 64.f);
    }
    __syncthreads();
    float n1v = (tid < 64) ? (h1[tid] - red[0]) * rsqrtf(red[1] + 1e-5f) * g1[tid] + be1[tid]
                           : 0.f;
    __syncthreads();
    if (tid < 64) h1[tid] = n1v;
    __syncthreads();
    if (tid < NCLS) {
      float acc = b2[tid];
      for (int c = 0; c < 64; ++c) acc += h1[c] * W2[c * NCLS + tid];
      out[b * NCLS + tid] = acc;
    }
    __syncthreads();
  }
}

// ---------------------------------------------------------------- launcher
struct HgFeat { float* sf; _Float16* sh; float* vf; _Float16* vh; };

// jax pytree flatten order (dict keys sorted alphabetically):
// gttfn params: W00 W01 W1 W10 W11 W2 Wg b1 b2 bg
enum { G_W00 = 0, G_W01, G_W1, G_W10, G_W11, G_W2, G_Wg, G_b1, G_b2, G_bg };

extern "C" void kernel_launch(void* const* d_in, const int* in_sizes, int n_in,
                              void* d_out, int out_size, void* d_ws, size_t ws_size,
                              hipStream_t stream) {
  (void)in_sizes; (void)n_in; (void)out_size; (void)ws_size;
  const int NT0 = NB * N0, NT1 = NB * M1, NT2 = NB * M2;
  const int IL0 = 0, IL1 = 10, IMX = 20, PMX0 = 22, PMX1 = 24, RHO = 26;
  const int SL00 = 36, SL01 = 46, SL10 = 56, SL11 = 66;
  const int SMX0 = 76, SMX1 = 78, IPOS = 80;
  const float* pos = (const float*)d_in[IPOS];

  char* wsp = (char*)d_ws;
  size_t off = 0;
  auto carve = [&](size_t bytes) -> void* {
    void* p = wsp + off;
    off += (bytes + 255) & ~(size_t)255;
    return p;
  };

  // --- packed fragment blobs
  _Float16* wblob[6];
  const int gbase[6] = {IL0, IL1, SL00, SL01, SL10, SL11};
  const int gcin[6] = {1, 32, 32, 32, 32, 32};
  for (int L = 0; L < 6; ++L) wblob[L] = (_Float16*)carve(WBLOB_B);
  _Float16* mblob[5];
  const int mxb[5] = {IMX, PMX0, PMX1, SMX0, SMX1};
  for (int i = 0; i < 5; ++i) mblob[i] = (_Float16*)carve(4 * FRAG_H * 2);

  // --- geometry buffers (sized for level 0, reused by later levels)
  _Float16* rbfb = (_Float16*)carve((size_t)NT0 * KNBR * NRBF * 2);
  float* rhatb = (float*)carve((size_t)NT0 * KNBR * 3 * 4);
  int* nbrb = (int*)carve((size_t)NT0 * KNBR * 4);

  auto mkfeat = [&](int NT) {
    HgFeat f;
    f.sf = (float*)carve((size_t)NT * CCH * 4);
    f.sh = (_Float16*)carve((size_t)NT * CCH * 2);
    f.vf = (float*)carve((size_t)3 * NT * CCH * 4);
    f.vh = (_Float16*)carve((size_t)3 * NT * CCH * 2);
    return f;
  };
  HgFeat X = mkfeat(NT0), Y = mkfeat(NT0);  // ping-pong, reused at all levels

  float* cents0 = (float*)carve((size_t)NB * M1 * 3 * 4);
  float* cents1 = (float*)carve((size_t)NB * M2 * 3 * 4);
  int* bidx0 = (int*)carve((size_t)NB * M1 * KNBR * 4);
  int* bidx1 = (int*)carve((size_t)NB * M2 * KNBR * 4);
  float* desc = (float*)carve((size_t)NB * 2 * CCH * 4);

  // --- pack weights into B-fragment blobs (once per layer)
  auto packb = [&](const float* W, int vr, int ldw, int k0, int nb, _Float16* dst) {
    hg_pack_b<<<(nb * FRAG_H + 255) / 256, 256, 0, stream>>>(W, vr, ldw, k0, nb, dst);
  };
  for (int L = 0; L < 6; ++L) {
    int pb = gbase[L];
    const float* W1 = (const float*)d_in[pb + G_W1];
    const float* W2 = (const float*)d_in[pb + G_W2];
    packb(W1, 32, RH, 0, 4, wblob[L] + 0 * FRAG_H);
    packb(W2, 64, 4 * CCH, 0, 8, wblob[L] + 4 * FRAG_H);
    packb(W2, 64, 4 * CCH, 32, 8, wblob[L] + 12 * FRAG_H);
    packb((const float*)d_in[pb + G_W00], gcin[L], CCH, 0, 2, wblob[L] + 20 * FRAG_H);
    packb((const float*)d_in[pb + G_W11], gcin[L], CCH, 0, 2, wblob[L] + 22 * FRAG_H);
    packb((const float*)d_in[pb + G_W01], gcin[L], CCH, 0, 2, wblob[L] + 24 * FRAG_H);
    packb((const float*)d_in[pb + G_W10], gcin[L], CCH, 0, 2, wblob[L] + 26 * FRAG_H);
  }
  for (int i = 0; i < 5; ++i) {
    packb((const float*)d_in[mxb[i] + 0], 32, CCH, 0, 2, mblob[i] + 0 * FRAG_H);
    packb((const float*)d_in[mxb[i] + 1], 32, CCH, 0, 2, mblob[i] + 2 * FRAG_H);
  }

  auto run_layer = [&](int NT, int res, const HgFeat& in, const HgFeat& outF, int L) {
    int pb = gbase[L];
    hg_gttfn<<<NT / 8, 256, 0, stream>>>(
        NT, res, in.sh, in.sf, in.vh, in.vf, rbfb, rhatb, nbrb, wblob[L],
        (const float*)d_in[pb + G_b1], (const float*)d_in[pb + G_b2],
        (const float*)d_in[pb + G_Wg], (const float*)d_in[pb + G_bg],
        outF.sf, outF.sh, outF.vf, outF.vh);
  };
  auto run_mixer = [&](int NT, const HgFeat& in, const HgFeat& outF, int mi) {
    hg_mixer<<<NT / 128, 256, 0, stream>>>(NT, in.sh, in.vh, mblob[mi],
                                           outF.sf, outF.sh, outF.vf, outF.vh);
  };

  // --- level 0 (N = 2048)
  hg_init_embed<<<NT0 / 256, 256, 0, stream>>>(pos, NT0, X.sf, X.sh, X.vf, X.vh);
  hg_knn<<<NT0 / 256, 256, 0, stream>>>(pos, N0, NT0, rbfb, rhatb, nbrb);
  run_layer(NT0, 0, X, Y, 0);
  run_layer(NT0, 1, Y, X, 1);
  run_mixer(NT0, X, Y, 0);  // level-0 features in Y

  // --- stage 0 (2048 -> 512, r = 0.2)
  hg_fps<<<NB, 256, 0, stream>>>(pos, N0, M1, cents0);
  hg_ballq<<<(NB * M1 + 255) / 256, 256, 0, stream>>>(pos, N0, cents0, M1,
                                                      NB * M1, 0.04f, bidx0);
  hg_pool<<<NT1 * CCH / 256, 256, 0, stream>>>(bidx0, NT1, Y.sf, Y.vf,
                                               (size_t)NT0 * CCH, X.sf, X.sh,
                                               X.vf, X.vh, (size_t)NT1 * CCH);
  run_mixer(NT1, X, Y, 1);
  hg_knn<<<NT1 / 256, 256, 0, stream>>>(cents0, M1, NT1, rbfb, rhatb, nbrb);
  run_layer(NT1, 1, Y, X, 2);
  run_layer(NT1, 1, X, Y, 3);
  run_mixer(NT1, Y, X, 3);  // stage-0 features in X

  // --- stage 1 (512 -> 128, r = 0.4)
  hg_fps<<<NB, 256, 0, stream>>>(cents0, M1, M2, cents1);
  hg_ballq<<<(NB * M2 + 255) / 256, 256, 0, stream>>>(cents0, M1, cents1, M2,
                                                      NB * M2, 0.16f, bidx1);
  hg_pool<<<NT2 * CCH / 256, 256, 0, stream>>>(bidx1, NT2, X.sf, X.vf,
                                               (size_t)NT1 * CCH, Y.sf, Y.sh,
                                               Y.vf, Y.vh, (size_t)NT2 * CCH);
  run_mixer(NT2, Y, X, 2);
  hg_knn<<<NT2 / 256, 256, 0, stream>>>(cents1, M2, NT2, rbfb, rhatb, nbrb);
  run_layer(NT2, 1, X, Y, 4);
  run_layer(NT2, 1, Y, X, 5);
  run_mixer(NT2, X, Y, 4);  // final features in Y

  // --- readout + head
  hg_desc<<<NB, 64, 0, stream>>>(Y.sf, Y.vf, (size_t)NT2 * CCH, M2, desc);
  hg_head<<<1, 128, 0, stream>>>(
      desc, (const float*)d_in[RHO + 0], (const float*)d_in[RHO + 3],
      (const float*)d_in[RHO + 8], (const float*)d_in[RHO + 6],
      (const float*)d_in[RHO + 1], (const float*)d_in[RHO + 4],
      (const float*)d_in[RHO + 9], (const float*)d_in[RHO + 7],
      (const float*)d_in[RHO + 2], (const float*)d_in[RHO + 5], (float*)d_out);
}
